// Attention_36155034697810
// MI455X (gfx1250) — matrix operand order repaired and checked by reading, not compiled, not run on hardware
//
#include <hip/hip_runtime.h>

// ---------------------------------------------------------------------------
// MI455X (gfx1250, wave32) flash-attention implementation of SAM-ViT global
// attention with decomposed relative position bias.
// All large matmuls use v_wmma_f32_16x16x32_f16 (f16 in, fp32 accumulate).
// Attention computes S^T = K.Q^T so softmax reductions are per-lane (one
// query per lane) and the P^T redistribution is a single half-wave exchange:
// no LDS traffic in the flash loop at all.
// ---------------------------------------------------------------------------

typedef __attribute__((ext_vector_type(16))) _Float16 v16h;
typedef __attribute__((ext_vector_type(8)))  _Float16 v8h;
typedef __attribute__((ext_vector_type(8)))  float    v8f;

#define NH   12
#define S_   4096
#define C_   768
#define HD   64

__device__ __forceinline__ v8f wmma16(v16h a, v16h b, v8f c) {
  // D = A(16x32) * B(32x16) + C, fp32 accumulate
  return __builtin_amdgcn_wmma_f32_16x16x32_f16(false, a, false, b,
                                                (short)0, c, false, false);
}

// A-fragment (16x32 f16): lane m (0..15) = row M; this lane's 16 K-elements
// are kstep*32 + (lane>>4)*16 + 0..15, with base = kstep*32 + (lane>>4)*8.
__device__ __forceinline__ v16h load_a_from_f16(const _Float16* row, int base) {
  v8h lo = *(const v8h*)(row + base + (base & 8));
  v8h hi = *(const v8h*)(row + base + (base & 8) + 8);
  v16h a;
#pragma unroll
  for (int j = 0; j < 8; ++j) { a[j] = lo[j]; a[8 + j] = hi[j]; }
  return a;
}

// ---------------------------------------------------------------------------
// K0a: transpose fp32 (K x N) weight -> f16 (N x K): WMMA B-fragments become
// contiguous 32B loads.
// ---------------------------------------------------------------------------
__global__ void transpose_to_f16(const float* __restrict__ w,
                                 _Float16* __restrict__ wT, int K, int N) {
  __shared__ float tile[32][33];
  int n0 = blockIdx.x * 32, k0 = blockIdx.y * 32;
  int tx = threadIdx.x, ty = threadIdx.y;  // 32 x 8
#pragma unroll
  for (int yy = 0; yy < 4; ++yy)
    tile[ty * 4 + yy][tx] = w[(size_t)(k0 + ty * 4 + yy) * N + n0 + tx];
  __syncthreads();
#pragma unroll
  for (int yy = 0; yy < 4; ++yy)
    wT[(size_t)(n0 + ty * 4 + yy) * K + k0 + tx] =
        (_Float16)tile[tx][ty * 4 + yy];
}

// ---------------------------------------------------------------------------
// K0b: bulk fp32 -> f16 convert (x), so the QKV GEMM inner loop has no cvt
// chains -- A-fragments become two 16B loads.
// ---------------------------------------------------------------------------
__global__ void convert_to_f16(const float* __restrict__ in,
                               _Float16* __restrict__ out, int n8) {
  int i = blockIdx.x * blockDim.x + threadIdx.x;
  if (i >= n8) return;
  const float4* p = (const float4*)(in + (size_t)i * 8);
  float4 f0 = p[0], f1 = p[1];
  v8h o;
  o[0]=(_Float16)f0.x; o[1]=(_Float16)f0.y; o[2]=(_Float16)f0.z; o[3]=(_Float16)f0.w;
  o[4]=(_Float16)f1.x; o[5]=(_Float16)f1.y; o[6]=(_Float16)f1.z; o[7]=(_Float16)f1.w;
  *(v8h*)(out + (size_t)i * 8) = o;
}

// ---------------------------------------------------------------------------
// K1: QKV = x(4096x768,f16) @ qkv_w + b, scattered to per-head
// q,k (f16 row-major [head][s][c]) and v (f16 transposed [head][c][s]).
// Grid (64 Mtiles, 36 Ntiles), 128 threads = 4 waves, 16-row strip per wave.
// ---------------------------------------------------------------------------
__global__ __launch_bounds__(128) void qkv_gemm(
    const _Float16* __restrict__ xh, const _Float16* __restrict__ wT,
    const float* __restrict__ bias, _Float16* __restrict__ qh_,
    _Float16* __restrict__ kf, _Float16* __restrict__ vT) {
  int Mt = blockIdx.x, Nt = blockIdx.y;
  int wave = threadIdx.x >> 5, lane = threadIdx.x & 31;
  int ln = lane & 15, hgrp = lane >> 4;
  int mstrip = Mt * 64 + wave * 16;
  const _Float16* arow = xh + (size_t)(mstrip + ln) * C_;

  v8f acc[4];
  v8f zero = {0.f, 0.f, 0.f, 0.f, 0.f, 0.f, 0.f, 0.f};
#pragma unroll
  for (int nt = 0; nt < 4; ++nt) acc[nt] = zero;

  for (int kc = 0; kc < 12; ++kc) {
#pragma unroll
    for (int ks = 0; ks < 2; ++ks) {
      v16h a = load_a_from_f16(arow, kc * 64 + ks * 32 + hgrp * 8);
      int koff = kc * 64 + ks * 32 + hgrp * 16;
#pragma unroll
      for (int nt = 0; nt < 4; ++nt) {
        v16h b = *(const v16h*)(wT + (size_t)(Nt * 64 + nt * 16 + ln) * C_ + koff);
        acc[nt] = wmma16(a, b, acc[nt]);
      }
    }
  }

  int which = Nt / NH, head = Nt % NH;  // 64-col tiles align with (q|k|v, head)
  int sbase = mstrip + 8 * hgrp;
#pragma unroll
  for (int nt = 0; nt < 4; ++nt) {
    int c = nt * 16 + ln;                 // channel within head
    float bv = bias[Nt * 64 + c];
    if (which < 2) {                      // q or k: row-major [head][s][c]
      _Float16* op = (which == 0 ? qh_ : kf) + ((size_t)head * S_ + sbase) * HD + c;
#pragma unroll
      for (int i = 0; i < 8; ++i) op[i * HD] = (_Float16)(acc[nt][i] + bv);
    } else {                              // v: transposed [head][c][s]
      v8h pk;
#pragma unroll
      for (int i = 0; i < 8; ++i) pk[i] = (_Float16)(acc[nt][i] + bv);
      *(v8h*)(vT + ((size_t)head * HD + c) * S_ + sbase) = pk;  // contiguous s
    }
  }
}

// ---------------------------------------------------------------------------
// K2: decomposed rel-pos bias tables (tiny: 0.8 GF total, plain VALU).
// mode 0: relh[head][qh][qw][kh] = sum_c q * rel_pos_h[qh-kh+63][c]
// mode 1: relw[head][qh][qw][kw] = sum_c q * rel_pos_w[qw-kw+63][c]
// (q already rounded to f16 -- consistent with the matmul path precision)
// ---------------------------------------------------------------------------
__global__ void relbias_kernel(const _Float16* __restrict__ qh_,
                               const float* __restrict__ tabh,
                               const float* __restrict__ tabw,
                               float* __restrict__ relh,
                               float* __restrict__ relw) {
  int qhx = blockIdx.x, head = blockIdx.y, mode = blockIdx.z;
  const _Float16* qbase = qh_ + ((size_t)head * S_ + qhx * 64) * HD;
  float* out = (mode == 0 ? relh : relw);
  for (int t = 0; t < 16; ++t) {
    int idx = threadIdx.x + t * 256;  // 0..4095
    int qw = idx >> 6, kk = idx & 63;
    const _Float16* qrow = qbase + qw * HD;
    const float* tab = (mode == 0) ? (tabh + (size_t)(qhx - kk + 63) * HD)
                                   : (tabw + (size_t)(qw - kk + 63) * HD);
    float sum = 0.f;
#pragma unroll 8
    for (int c = 0; c < HD; ++c) sum += (float)qrow[c] * tab[c];
    out[(((size_t)head * 64 + qhx) * 64 + qw) * 64 + kk] = sum;
  }
}

// ---------------------------------------------------------------------------
// K3: flash attention, transposed formulation. Grid (64 q-tiles, 12 heads),
// 128 threads = 4 waves; each wave owns 16 queries (query = lane&15).
// Per 64-key tile: 8 WMMA  S^T = K . Q^T  (A = K-tile, B = Q, persistent),
// per-lane softmax (one shfl_xor(16) per reduction), half-wave exchange to
// build P^T B-fragments, 8 WMMA  O^T += V^T . P^T.  No LDS.
// ---------------------------------------------------------------------------
__global__ __launch_bounds__(128) void attn_kernel(
    const _Float16* __restrict__ qh_, const _Float16* __restrict__ kf,
    const _Float16* __restrict__ vT, const float* __restrict__ relh,
    const float* __restrict__ relw, _Float16* __restrict__ of) {
  int qt = blockIdx.x, head = blockIdx.y;
  int wave = threadIdx.x >> 5, lane = threadIdx.x & 31;
  int ln = lane & 15, hgrp = lane >> 4;

  const float scale = 0.125f;             // hd^-0.5 = 1/8
  int qrowidx = qt * 64 + wave * 16 + ln; // this lane's query s-index
  int qw = wave * 16 + ln;                // query w-coordinate (qh == qt)

  // Persistent Q B-fragments: b[j] = Q[c = ks*32 + hgrp*16 + j][query]
  const _Float16* qrow = qh_ + ((size_t)head * S_ + qrowidx) * HD;
  v16h qb0 = *(const v16h*)(qrow + 0 * 32 + hgrp * 16);
  v16h qb1 = *(const v16h*)(qrow + 1 * 32 + hgrp * 16);

  const float* relh_base = relh + (((size_t)head * 64 + qt) * 64) * 64;
  const float* relw_base = relw + (((size_t)head * 64 + qt) * 64) * 64;

  // rel_w bias: key-tile invariant, kw = mt*16 + 8*hgrp + i
  float bw[4][8];
#pragma unroll
  for (int mt = 0; mt < 4; ++mt) {
    const float* p = relw_base + (size_t)qw * 64 + mt * 16 + 8 * hgrp;
#pragma unroll
    for (int i = 0; i < 8; ++i) bw[mt][i] = p[i];
  }

  v8f o[4];
  v8f zero = {0.f, 0.f, 0.f, 0.f, 0.f, 0.f, 0.f, 0.f};
#pragma unroll
  for (int ct = 0; ct < 4; ++ct) o[ct] = zero;
  float mrun = -1e30f, lrun = 0.f;

  const _Float16* kbase = kf + (size_t)head * S_ * HD;
  const _Float16* vbase = vT + (size_t)head * HD * S_;

  for (int t = 0; t < 64; ++t) {
    if (t < 63)
      __builtin_prefetch(kbase + (size_t)(t + 1) * 64 * HD, 0, 3);

    // --- S^T = K . Q^T : D[key = mt*16+8*hgrp+i][query = ln] -------------
    v8f s[4];
#pragma unroll
    for (int mt = 0; mt < 4; ++mt) s[mt] = zero;
#pragma unroll
    for (int mt = 0; mt < 4; ++mt) {
      const _Float16* krow = kbase + (size_t)(t * 64 + mt * 16 + ln) * HD;
      s[mt] = wmma16(load_a_from_f16(krow, 0 * 32 + hgrp * 8), qb0, s[mt]);
      s[mt] = wmma16(load_a_from_f16(krow, 1 * 32 + hgrp * 8), qb1, s[mt]);
    }

    // --- scale + bias, per-lane online softmax ---------------------------
    float bh = relh_base[(size_t)qw * 64 + t];  // kh == t for this key tile
    float lg[4][8];
    float mx = -1e30f;
#pragma unroll
    for (int mt = 0; mt < 4; ++mt)
#pragma unroll
      for (int i = 0; i < 8; ++i) {
        float v = fmaf(s[mt][i], scale, bh + bw[mt][i]);
        lg[mt][i] = v;
        mx = fmaxf(mx, v);
      }
    mx = fmaxf(mx, __shfl_xor(mx, 16));     // other 32 keys of this query
    float newm = fmaxf(mrun, mx);
    float cf = __expf(mrun - newm);
    mrun = newm;
    float rs = 0.f;
#pragma unroll
    for (int mt = 0; mt < 4; ++mt)
#pragma unroll
      for (int i = 0; i < 8; ++i) {
        float p = __expf(lg[mt][i] - newm);
        lg[mt][i] = p;
        rs += p;
      }
    rs += __shfl_xor(rs, 16);
    lrun = lrun * cf + rs;
#pragma unroll
    for (int ct = 0; ct < 4; ++ct)
#pragma unroll
      for (int i = 0; i < 8; ++i) o[ct][i] *= cf;

    // --- build P^T B-fragments via half-wave exchange --------------------
    // pb[ks][j] = P[key_local = ks*32 + hgrp*16 + j][query ln]; source D
    // element: mt = key_local>>4, srcHalf = j>>3, i = j&7 -> lane ln+16*srcHalf.
    v16h pb0, pb1;
#pragma unroll
    for (int ks = 0; ks < 2; ++ks) {
#pragma unroll
      for (int i = 0; i < 8; ++i) {
        float a0 = lg[2 * ks + 0][i];          // this lane's mt = 2ks (half 0 dest)
        float a1 = lg[2 * ks + 1][i];          // this lane's mt = 2ks+1 (half 1 dest)
        float send = hgrp ? a0 : a1;           // what the partner lane needs
        float recv = __shfl_xor(send, 16);
        float own  = hgrp ? a1 : a0;           // this lane's own target mt
        float lo = hgrp ? recv : own;          // j<8  : from half 0
        float hi = hgrp ? own  : recv;         // j>=8 : from half 1
        if (ks == 0) { pb0[i] = (_Float16)lo; pb0[8 + i] = (_Float16)hi; }
        else         { pb1[i] = (_Float16)lo; pb1[8 + i] = (_Float16)hi; }
      }
    }

    // --- O^T += V^T . P^T : D[c = ct*16+8*hgrp+i][query = ln] ------------
#pragma unroll
    for (int ct = 0; ct < 4; ++ct) {
      const _Float16* vrow = vbase + (size_t)(ct * 16 + ln) * S_ + t * 64;
      o[ct] = wmma16(load_a_from_f16(vrow, 0 * 32 + hgrp * 8), pb0, o[ct]);
      o[ct] = wmma16(load_a_from_f16(vrow, 1 * 32 + hgrp * 8), pb1, o[ct]);
    }
  }

  // --- epilogue: normalize, write f16 (S, C); c contiguous per lane -------
  float inv = 1.f / lrun;
#pragma unroll
  for (int ct = 0; ct < 4; ++ct) {
    v8h pk;
#pragma unroll
    for (int i = 0; i < 8; ++i) pk[i] = (_Float16)(o[ct][i] * inv);
    *(v8h*)(of + (size_t)qrowidx * C_ + head * HD + ct * 16 + hgrp * 8) = pk;
  }
}

// ---------------------------------------------------------------------------
// K4: out = attn_out(4096x768 f16) @ proj_w + proj_b   (fp32 result)
// ---------------------------------------------------------------------------
__global__ __launch_bounds__(128) void proj_gemm(
    const _Float16* __restrict__ of, const _Float16* __restrict__ wT,
    const float* __restrict__ bias, float* __restrict__ out) {
  int Mt = blockIdx.x, Nt = blockIdx.y;  // 64 x 12
  int wave = threadIdx.x >> 5, lane = threadIdx.x & 31;
  int ln = lane & 15, hgrp = lane >> 4;
  int mstrip = Mt * 64 + wave * 16;
  const _Float16* arow = of + (size_t)(mstrip + ln) * C_;

  v8f acc[4];
  v8f zero = {0.f, 0.f, 0.f, 0.f, 0.f, 0.f, 0.f, 0.f};
#pragma unroll
  for (int nt = 0; nt < 4; ++nt) acc[nt] = zero;

  for (int kc = 0; kc < 12; ++kc) {
#pragma unroll
    for (int ks = 0; ks < 2; ++ks) {
      v16h a = load_a_from_f16(arow, kc * 64 + ks * 32 + hgrp * 8);
      int koff = kc * 64 + ks * 32 + hgrp * 16;
#pragma unroll
      for (int nt = 0; nt < 4; ++nt) {
        v16h b = *(const v16h*)(wT + (size_t)(Nt * 64 + nt * 16 + ln) * C_ + koff);
        acc[nt] = wmma16(a, b, acc[nt]);
      }
    }
  }
#pragma unroll
  for (int nt = 0; nt < 4; ++nt) {
    int col = Nt * 64 + nt * 16 + ln;
    float bv = bias[col];
    float* op = out + (size_t)(mstrip + 8 * hgrp) * C_ + col;
#pragma unroll
    for (int i = 0; i < 8; ++i) op[i * C_] = acc[nt][i] + bv;
  }
}

// ---------------------------------------------------------------------------
extern "C" void kernel_launch(void* const* d_in, const int* in_sizes, int n_in,
                              void* d_out, int out_size, void* d_ws,
                              size_t ws_size, hipStream_t stream) {
  const float* x      = (const float*)d_in[0];  // (1,64,64,768)
  const float* qkv_w  = (const float*)d_in[1];  // (768, 2304)
  const float* qkv_b  = (const float*)d_in[2];  // (2304,)
  const float* proj_w = (const float*)d_in[3];  // (768, 768)
  const float* proj_b = (const float*)d_in[4];  // (768,)
  const float* rph    = (const float*)d_in[5];  // (127, 64)
  const float* rpw    = (const float*)d_in[6];  // (127, 64)
  float* out = (float*)d_out;                   // (4096, 768) fp32

  // workspace layout (~61 MB, 256B-aligned chunks)
  char* ws = (char*)d_ws;
  size_t off = 0;
  auto alloc = [&](size_t bytes) -> void* {
    void* p = ws + off;
    off += (bytes + 255) & ~(size_t)255;
    return p;
  };
  _Float16* xh   = (_Float16*)alloc((size_t)S_ * C_ * 2);        // 6.3 MB
  _Float16* qh_  = (_Float16*)alloc((size_t)NH * S_ * HD * 2);   // 6.3 MB
  _Float16* kf   = (_Float16*)alloc((size_t)NH * S_ * HD * 2);   // 6.3 MB
  _Float16* vT   = (_Float16*)alloc((size_t)NH * HD * S_ * 2);   // 6.3 MB
  float*    relh = (float*)alloc((size_t)NH * 64 * 64 * 64 * 4); // 12.6 MB
  float*    relw = (float*)alloc((size_t)NH * 64 * 64 * 64 * 4); // 12.6 MB
  _Float16* of   = (_Float16*)alloc((size_t)S_ * C_ * 2);        // 6.3 MB
  _Float16* wTq  = (_Float16*)alloc((size_t)3 * C_ * C_ * 2);    // 3.5 MB
  _Float16* wTp  = (_Float16*)alloc((size_t)C_ * C_ * 2);        // 1.2 MB

  // K0: weight transposes + x conversion -> f16
  transpose_to_f16<<<dim3(3 * C_ / 32, C_ / 32), dim3(32, 8), 0, stream>>>(
      qkv_w, wTq, C_, 3 * C_);
  transpose_to_f16<<<dim3(C_ / 32, C_ / 32), dim3(32, 8), 0, stream>>>(
      proj_w, wTp, C_, C_);
  {
    int n8 = S_ * C_ / 8;
    convert_to_f16<<<(n8 + 255) / 256, 256, 0, stream>>>(x, xh, n8);
  }

  // K1: QKV projection + scatter
  qkv_gemm<<<dim3(S_ / 64, 3 * NH), 128, 0, stream>>>(xh, wTq, qkv_b,
                                                      qh_, kf, vT);

  // K2: rel-pos bias tables
  relbias_kernel<<<dim3(64, NH, 2), 256, 0, stream>>>(qh_, rph, rpw,
                                                      relh, relw);

  // K3: flash attention (no-LDS transposed formulation)
  attn_kernel<<<dim3(S_ / 64, NH), 128, 0, stream>>>(qh_, kf, vT,
                                                     relh, relw, of);

  // K4: output projection
  proj_gemm<<<dim3(S_ / 64, C_ / 64), 128, 0, stream>>>(of, wTp, proj_b, out);
}
